// shiftmlp_1013612281894
// MI455X (gfx1250) — compile-verified, matrix operand order checked by reading, not run on hardware
//
#include <hip/hip_runtime.h>
#include <cstdint>
#include <cstddef>

// CDNA5 wave32 WMMA types
typedef __attribute__((ext_vector_type(16))) __bf16 v16bf;
typedef __attribute__((ext_vector_type(8)))  float  v8f;

// POD vector structs (bit_cast-safe)
struct U4 { uint32_t x, y, z, w; };
struct F4 { float x, y, z, w; };
struct U2 { uint32_t x, y; };
struct U4x2 { U4 lo, hi; };

typedef __attribute__((address_space(3))) const void* lds_cptr_t;

__device__ __forceinline__ uint16_t f2bf(float f) {
  uint32_t u = __float_as_uint(f);
  u += 0x7FFFu + ((u >> 16) & 1u);          // round-to-nearest-even
  return (uint16_t)(u >> 16);
}
__device__ __forceinline__ float bf2f(uint32_t hbits) {
  return __uint_as_float(hbits << 16);
}
__device__ __forceinline__ uint32_t pk2(uint16_t a, uint16_t b) {
  return (uint32_t)a | ((uint32_t)b << 16);
}
// LDS byte offset of a shared-memory object (generic -> AS3 -> offset)
__device__ __forceinline__ uint32_t lds_addr_of(const void* p) {
  return (uint32_t)(uintptr_t)(lds_cptr_t)p;
}

constexpr int kH = 128, kW = 128;
constexpr int kRows = 8 * kH * kW;    // B*H*W = 131072 GEMM rows
constexpr int AST = 40;               // LDS row strides (ushorts), 16B-aligned groups
constexpr int BST = 40;

// One-shot fp32 -> bf16 weight conversion (row-major 256x256), 8 elems/thread.
__global__ void __launch_bounds__(256)
cvt_w(const float* __restrict__ w, uint16_t* __restrict__ o) {
  const int i = (blockIdx.x * 256 + threadIdx.x) * 8;
  const F4 a = *reinterpret_cast<const F4*>(w + i);
  const F4 b = *reinterpret_cast<const F4*>(w + i + 4);
  const U4 pk{ pk2(f2bf(a.x), f2bf(a.y)), pk2(f2bf(a.z), f2bf(a.w)),
               pk2(f2bf(b.x), f2bf(b.y)), pk2(f2bf(b.z), f2bf(b.w)) };
  *reinterpret_cast<U4*>(o + i) = pk;
}

// Fused shift + GEMM (M=131072, N=256, K=256), D = shift(A) * W^T + bias.
// FIRST: A = x (fp32, shift along H), out bf16.  !FIRST: A = t (bf16, shift along W), out fp32.
// Double-buffered LDS + single barrier per K-step: stage tile k+1 (async B copy
// via global_load_async_to_lds_b128 + VALU A gather) while WMMAs consume tile k.
template<bool FIRST>
__global__ void __launch_bounds__(256)
gemm_shift(const void* __restrict__ a_src, const uint16_t* __restrict__ wbf,
           const float* __restrict__ bias, void* __restrict__ dst)
{
  __shared__ alignas(16) uint16_t lds_a[2][64 * AST];    //  64 x 32 bf16 A tiles
  __shared__ alignas(16) uint16_t lds_b[2][256 * BST];   // 256 x 32 bf16 W tiles

  const int tid     = threadIdx.x;
  const int rowBase = blockIdx.x * 64;
  const int lane    = tid & 31;
  const int wave    = tid >> 5;
  const int l16     = lane & 15;
  const int khalf   = (lane >> 4) * 8;   // K sub-offset per WMMA 16-bit layout
  const int mw      = wave & 3;          // 16-row M slice within 64
  const int nw      = wave >> 2;         // 128-col N half within 256

  const v8f vzero = {0.f, 0.f, 0.f, 0.f, 0.f, 0.f, 0.f, 0.f};
  v8f acc[8];
#pragma unroll
  for (int i = 0; i < 8; ++i) acc[i] = vzero;

  // A-tile loader assignment: thread -> (row 0..63, 8-channel group)
  const int a_row = tid >> 2;
  const int a_cg  = (tid & 3) * 8;
  const int gr  = rowBase + a_row;
  const int gb  = gr >> 14;              // batch
  const int ghw = gr & 16383;
  const int gh  = ghw >> 7;
  const int gw  = ghw & 127;
  const size_t imgBase = ((size_t)gb << 14) * 256;

  // ---- stage W tile (async): 256 rows x 32 bf16 = 1024 16B chunks, 4/thread ----
  auto stageB = [&](int k0, int buf) {
#pragma unroll
    for (int g = 0; g < 4; ++g) {
      const int f  = g * 256 + tid;
      const int n  = f >> 2;
      const int gg = f & 3;
      const uint32_t la = lds_addr_of(&lds_b[buf][n * BST + gg * 8]);
      const uint64_t ga = (uint64_t)(uintptr_t)(wbf + (size_t)n * 256 + k0 + gg * 8);
      asm volatile("global_load_async_to_lds_b128 %0, %1, off"
                   :: "v"(la), "v"(ga) : "memory");
    }
  };

  // ---- stage A tile: shift-chunk boundaries (52/104/156/208) are multiples of
  //      4, so each aligned 4-channel subgroup has one shift; clamp + cndmask. ----
  auto stageA = [&](int k0, int buf) {
    uint32_t p[4];
#pragma unroll
    for (int sub = 0; sub < 2; ++sub) {
      const int c0 = k0 + a_cg + sub * 4;
      const int sh = (c0 >= 52) + (c0 >= 104) + (c0 >= 156) + (c0 >= 208) - 2;
      if (FIRST) {
        const int hs = gh - sh;
        const bool valid = (unsigned)hs < (unsigned)kH;
        const int hc = valid ? hs : gh;          // clamped, always in-bounds
        const F4 f4 = *reinterpret_cast<const F4*>(
            (const float*)a_src + imgBase + (((size_t)hc << 7) + gw) * 256 + c0);
        p[sub * 2 + 0] = valid ? pk2(f2bf(f4.x), f2bf(f4.y)) : 0u;
        p[sub * 2 + 1] = valid ? pk2(f2bf(f4.z), f2bf(f4.w)) : 0u;
      } else {
        const int wsrc = gw - sh;
        const bool valid = (unsigned)wsrc < (unsigned)kW;
        const int wc = valid ? wsrc : gw;
        const U2 d = *reinterpret_cast<const U2*>(
            (const uint16_t*)a_src + imgBase + (((size_t)gh << 7) + wc) * 256 + c0);
        p[sub * 2 + 0] = valid ? d.x : 0u;
        p[sub * 2 + 1] = valid ? d.y : 0u;
      }
    }
    const U4 pk{ p[0], p[1], p[2], p[3] };
    *reinterpret_cast<U4*>(&lds_a[buf][a_row * AST + a_cg]) = pk;
  };

  // ---- fragments + 8 WMMAs from buffer `buf` ----
  auto compute = [&](int buf) {
    U4x2 araw;
    const int abase = (mw * 16 + l16) * AST + khalf;
    araw.lo = *reinterpret_cast<const U4*>(&lds_a[buf][abase]);
    araw.hi = *reinterpret_cast<const U4*>(&lds_a[buf][abase + 16]);
    const v16bf av = __builtin_bit_cast(v16bf, araw);

    auto loadB = [&](int nt) -> v16bf {
      U4x2 braw;
      const int bbase = (nw * 128 + nt * 16 + l16) * BST + khalf;
      braw.lo = *reinterpret_cast<const U4*>(&lds_b[buf][bbase]);
      braw.hi = *reinterpret_cast<const U4*>(&lds_b[buf][bbase + 16]);
      return __builtin_bit_cast(v16bf, braw);
    };
    v16bf bcur = loadB(0);
#pragma unroll
    for (int nt = 0; nt < 8; ++nt) {
      const v16bf bnext = (nt < 7) ? loadB(nt + 1) : bcur;
      acc[nt] = __builtin_amdgcn_wmma_f32_16x16x32_bf16(
          /*neg_a=*/false, av, /*neg_b=*/false, bcur,
          /*c_mod=*/(short)0, acc[nt], /*reuse_a=*/false, /*reuse_b=*/false);
      bcur = bnext;
    }
  };

  // ---- pipelined main loop: one barrier per K-step ----
  stageB(0, 0);
  stageA(0, 0);
  asm volatile("s_wait_asynccnt 0x0" ::: "memory");
  __syncthreads();

  for (int k = 0; k < 7; ++k) {
    const int cur = k & 1;
    stageB((k + 1) * 32, cur ^ 1);   // overlaps the WMMA phase below
    stageA((k + 1) * 32, cur ^ 1);
    compute(cur);
    asm volatile("s_wait_asynccnt 0x0" ::: "memory");
    __syncthreads();
  }
  compute(1);                         // k = 7

  // ---- epilogue: C/D layout -> M = r + 8*(lane>=16), N = lane%16 ----
#pragma unroll
  for (int nt = 0; nt < 8; ++nt) {
    const int col = nw * 128 + nt * 16 + l16;
    const float bv = bias[col];
    const int mrow0 = rowBase + mw * 16 + (khalf ? 8 : 0);
#pragma unroll
    for (int r = 0; r < 8; ++r) {
      const float val = acc[nt][r] + bv;
      const size_t off = (size_t)(mrow0 + r) * 256 + col;
      if (FIRST) reinterpret_cast<uint16_t*>(dst)[off] = f2bf(val);
      else       reinterpret_cast<float*>(dst)[off]    = val;
    }
  }
}

// Depthwise 3x3 conv + bias + exact GELU, bf16 -> bf16, 4 channels/thread.
__global__ void __launch_bounds__(256)
dwconv_gelu(const uint16_t* __restrict__ y, const float* __restrict__ dww,
            const float* __restrict__ dwb, uint16_t* __restrict__ t)
{
  const size_t idx = (size_t)blockIdx.x * 256 + threadIdx.x;
  const size_t e = idx << 2;           // 4 consecutive channels
  const int c  = (int)(e & 255);
  const size_t r = e >> 8;
  const int b  = (int)(r >> 14);
  const int hw = (int)(r & 16383);
  const int h  = hw >> 7;
  const int w  = hw & 127;
  const size_t base = ((size_t)b << 14) * 256;

  float wk[4][9];
#pragma unroll
  for (int j = 0; j < 4; ++j)
#pragma unroll
    for (int k = 0; k < 9; ++k)
      wk[j][k] = dww[(size_t)(c + j) * 9 + k];

  float accv[4] = {0.f, 0.f, 0.f, 0.f};
#pragma unroll
  for (int dy = -1; dy <= 1; ++dy) {
    const int hy = h + dy;
    if ((unsigned)hy >= (unsigned)kH) continue;
#pragma unroll
    for (int dx = -1; dx <= 1; ++dx) {
      const int wx = w + dx;
      if ((unsigned)wx >= (unsigned)kW) continue;
      const U2 d = *reinterpret_cast<const U2*>(
          y + base + (((size_t)hy << 7) + wx) * 256 + c);
      const int ki = (dy + 1) * 3 + (dx + 1);
      accv[0] += bf2f(d.x & 0xFFFFu) * wk[0][ki];
      accv[1] += bf2f(d.x >> 16)     * wk[1][ki];
      accv[2] += bf2f(d.y & 0xFFFFu) * wk[2][ki];
      accv[3] += bf2f(d.y >> 16)     * wk[3][ki];
    }
  }

  uint16_t o[4];
#pragma unroll
  for (int j = 0; j < 4; ++j) {
    const float v = accv[j] + dwb[c + j];
    const float g = 0.5f * v * (1.0f + erff(v * 0.70710678118654752f));
    o[j] = f2bf(g);
  }
  const U2 out{ pk2(o[0], o[1]), pk2(o[2], o[3]) };
  *reinterpret_cast<U2*>(t + e) = out;
}

extern "C" void kernel_launch(void* const* d_in, const int* in_sizes, int n_in,
                              void* d_out, int out_size, void* d_ws, size_t ws_size,
                              hipStream_t stream) {
  (void)in_sizes; (void)n_in; (void)out_size; (void)ws_size;
  const float* x   = (const float*)d_in[0];
  // d_in[1]=H, d_in[2]=W (scalar ints; fixed at 128 here)
  const float* w1  = (const float*)d_in[3];
  const float* b1  = (const float*)d_in[4];
  const float* dww = (const float*)d_in[5];
  const float* dwb = (const float*)d_in[6];
  const float* w2  = (const float*)d_in[7];
  const float* b2  = (const float*)d_in[8];

  // workspace layout: y (64 MB bf16) | t (64 MB bf16) | wbf1 (128 KB) | wbf2 (128 KB)
  uint16_t* y    = (uint16_t*)d_ws;
  uint16_t* t    = y + (size_t)kRows * 256;
  uint16_t* wbf1 = t + (size_t)kRows * 256;
  uint16_t* wbf2 = wbf1 + 256 * 256;

  cvt_w<<<32, 256, 0, stream>>>(w1, wbf1);
  cvt_w<<<32, 256, 0, stream>>>(w2, wbf2);
  gemm_shift<true ><<<kRows / 64, 256, 0, stream>>>((const void*)x, wbf1, b1, (void*)y);
  dwconv_gelu      <<<kRows / 4,  256, 0, stream>>>(y, dww, dwb, t);
  gemm_shift<false><<<kRows / 64, 256, 0, stream>>>((const void*)t, wbf2, b2, d_out);
}
